// BigramLanguageModel_32384053411918
// MI455X (gfx1250) — matrix operand order repaired and checked by reading, not compiled, not run on hardware
//
#include <hip/hip_runtime.h>
#include <hip/hip_bf16.h>

// ---------------- model constants ----------------
#define Bc   64
#define Tc   256
#define Cc   512
#define Hc   8
#define Lc   8
#define Vc   100
#define HDc  64
#define FFc  2048
#define BTc  (Bc * Tc)           // 16384 rows
#define VPADc 128                // LM head N padded to 128

typedef __bf16 bf16_t;
typedef __attribute__((ext_vector_type(16))) __bf16 v16bf;
typedef __attribute__((ext_vector_type(8)))  float  v8f;

union Frag { v16bf v; uint4 q[2]; };

__device__ __forceinline__ v8f wmma_bf16(const Frag& a, const Frag& b, v8f c) {
    return __builtin_amdgcn_wmma_f32_16x16x32_bf16(false, a.v, false, b.v,
                                                   (short)0, c, false, false);
}

// ---------------- embedding: x = tok_emb[idx] + pos_emb ----------------
__global__ __launch_bounds__(256) void embed_kernel(const int* __restrict__ idx,
                                                    const float* __restrict__ tok,
                                                    const float* __restrict__ pos,
                                                    float* __restrict__ x) {
    int bt = blockIdx.x;
    int tid = threadIdx.x;
    int token = idx[bt];
    int t = bt & (Tc - 1);
    const float* tr = tok + (size_t)token * Cc;
    const float* pr = pos + (size_t)t * Cc;
    float* xr = x + (size_t)bt * Cc;
    xr[tid]       = tr[tid]       + pr[tid];
    xr[tid + 256] = tr[tid + 256] + pr[tid + 256];
}

// ---------------- LayerNorm (f32 in, bf16 out) ----------------
__global__ __launch_bounds__(256) void ln_kernel(const float* __restrict__ x,
                                                 const float* __restrict__ g,
                                                 const float* __restrict__ b,
                                                 bf16_t* __restrict__ out) {
    int row = blockIdx.x, tid = threadIdx.x;
    const float* xr = x + (size_t)row * Cc;
    float a0 = xr[tid], a1 = xr[tid + 256];
    float s = a0 + a1;
#pragma unroll
    for (int d = 16; d; d >>= 1) s += __shfl_xor(s, d, 32);
    __shared__ float r1[8], r2[8];
    if ((tid & 31) == 0) r1[tid >> 5] = s;
    __syncthreads();
    float tot = 0.f;
#pragma unroll
    for (int i = 0; i < 8; ++i) tot += r1[i];
    float mean = tot * (1.f / Cc);
    float d0 = a0 - mean, d1 = a1 - mean;
    float q = d0 * d0 + d1 * d1;
#pragma unroll
    for (int d = 16; d; d >>= 1) q += __shfl_xor(q, d, 32);
    if ((tid & 31) == 0) r2[tid >> 5] = q;
    __syncthreads();
    float vt = 0.f;
#pragma unroll
    for (int i = 0; i < 8; ++i) vt += r2[i];
    float rstd = rsqrtf(vt * (1.f / Cc) + 1e-5f);
    bf16_t* orow = out + (size_t)row * Cc;
    orow[tid]       = (bf16_t)(d0 * rstd * g[tid]       + b[tid]);
    orow[tid + 256] = (bf16_t)(d1 * rstd * g[tid + 256] + b[tid + 256]);
}

// ---------------- weight conversion ----------------
// generic: W [K][Nsrc] f32  ->  Bt [Npad][K] bf16 (transposed, zero-padded rows)
__global__ __launch_bounds__(256) void conv_wt(const float* __restrict__ W,
                                               bf16_t* __restrict__ Bt,
                                               int K, int Nsrc, int Npad) {
    int i = blockIdx.x * 256 + threadIdx.x;         // over Npad*K
    int n = i / K, k = i - n * K;
    float v = (n < Nsrc) ? W[(size_t)k * Nsrc + n] : 0.f;
    Bt[i] = (bf16_t)v;
}

// Wq/Wk/Wv layer slice: [H][C][HD] f32 -> Bt [C][C] bf16 with col n = h*HD+d
__global__ __launch_bounds__(256) void conv_qkv(const float* __restrict__ W,
                                                bf16_t* __restrict__ Bt) {
    int i = blockIdx.x * 256 + threadIdx.x;         // over C*C
    int n = i >> 9, k = i & (Cc - 1);
    int h = n >> 6, d = n & (HDc - 1);
    Bt[i] = (bf16_t)W[((size_t)h * Cc + k) * HDc + d];
}

// f32 -> bf16 elementwise
__global__ __launch_bounds__(256) void cast_kernel(const float* __restrict__ in,
                                                   bf16_t* __restrict__ out) {
    int i = blockIdx.x * 256 + threadIdx.x;
    out[i] = (bf16_t)in[i];
}

// ---------------- generic bf16 GEMM: out = A[M][K] * Bt[N][K]^T (+bias, epilogue) ----------------
enum { EPI_BF16 = 0, EPI_BF16_RELU = 1, EPI_RES_F32 = 2, EPI_F32_MASK = 3 };

template <int MODE>
__global__ __launch_bounds__(256, 2) void gemm_bf16(const bf16_t* __restrict__ A,
                                                    const bf16_t* __restrict__ Bt,
                                                    const float* __restrict__ bias,
                                                    const float* __restrict__ res,
                                                    float* __restrict__ outF,
                                                    bf16_t* __restrict__ outB,
                                                    int M, int N, int K,
                                                    int ldOut, int Nvalid) {
    __shared__ bf16_t sA[128 * 40];                 // padded rows (40 halves) vs banks
    __shared__ bf16_t sB[128 * 40];
    const int tid = threadIdx.x, lane = tid & 31, wave = tid >> 5;
    const int wm = wave & 3, wn = wave >> 2;        // 4 (M) x 2 (N) waves
    const int m0 = blockIdx.y * 128, n0 = blockIdx.x * 128;
    const v8f vzero = {0.f, 0.f, 0.f, 0.f, 0.f, 0.f, 0.f, 0.f};
    v8f c[2][4];
#pragma unroll
    for (int i = 0; i < 2; ++i)
#pragma unroll
        for (int j = 0; j < 4; ++j) c[i][j] = vzero;

    const int kh = (lane >> 4) << 3;                // A-frag K half-select
    const int kb = (lane >> 4) << 3;                // B-frag K half-select

    for (int k0 = 0; k0 < K; k0 += 32) {
        __syncthreads();
#pragma unroll
        for (int r = 0; r < 2; ++r) {               // 512 16B-chunks per tile, 2/thread
            int ch = tid + r * 256;
            int row = ch >> 2, cg = (ch & 3) << 3;
            *(uint4*)(&sA[row * 40 + cg]) =
                *(const uint4*)(&A[(size_t)(m0 + row) * K + k0 + cg]);
            *(uint4*)(&sB[row * 40 + cg]) =
                *(const uint4*)(&Bt[(size_t)(n0 + row) * K + k0 + cg]);
        }
        if (k0 + 32 < K) {                          // prefetch next K-tile
            int ch = tid, row = ch >> 2, cg = (ch & 3) << 3;
            __builtin_prefetch(&A[(size_t)(m0 + row) * K + k0 + 32 + cg], 0, 1);
            __builtin_prefetch(&Bt[(size_t)(n0 + row) * K + k0 + 32 + cg], 0, 1);
        }
        __syncthreads();

        Frag a[2], b[4];
#pragma unroll
        for (int i = 0; i < 2; ++i) {
            int m = wm * 32 + i * 16 + (lane & 15);
            a[i].q[0] = *(const uint4*)(&sA[m * 40 + kh]);
            a[i].q[1] = *(const uint4*)(&sA[m * 40 + kh + 16]);
        }
#pragma unroll
        for (int j = 0; j < 4; ++j) {
            int n = wn * 64 + j * 16 + (lane & 15);
            b[j].q[0] = *(const uint4*)(&sB[n * 40 + kb]);
            b[j].q[1] = *(const uint4*)(&sB[n * 40 + kb + 16]);
        }
#pragma unroll
        for (int i = 0; i < 2; ++i)
#pragma unroll
            for (int j = 0; j < 4; ++j) c[i][j] = wmma_bf16(a[i], b[j], c[i][j]);
    }

    // epilogue: C/D layout -> row = r + (lane>>4)*8, col = lane&15
#pragma unroll
    for (int i = 0; i < 2; ++i)
#pragma unroll
        for (int j = 0; j < 4; ++j) {
            int n = n0 + wn * 64 + j * 16 + (lane & 15);
#pragma unroll
            for (int r = 0; r < 8; ++r) {
                int m = m0 + wm * 32 + i * 16 + r + ((lane >> 4) << 3);
                float v = c[i][j][r];
                if (MODE == EPI_F32_MASK) {
                    if (n < Nvalid) {
                        v += bias[n];
                        outF[(size_t)m * ldOut + n] = v;
                    }
                } else if (MODE == EPI_RES_F32) {
                    v += bias[n];
                    outF[(size_t)m * N + n] = res[(size_t)m * N + n] + v;
                } else {
                    if (bias) v += bias[n];
                    if (MODE == EPI_BF16_RELU) v = fmaxf(v, 0.f);
                    outB[(size_t)m * N + n] = (bf16_t)v;
                }
            }
        }
}

// ---------------- fused causal flash attention (per b,h, 128-row block) ----------------
__global__ __launch_bounds__(256, 1) void attn_kernel(const bf16_t* __restrict__ qg,
                                                      const bf16_t* __restrict__ kg,
                                                      const bf16_t* __restrict__ vg,
                                                      bf16_t* __restrict__ og) {
    __shared__ bf16_t sK[256 * 72];                 // k rows [s][d], padded
    __shared__ bf16_t sV[64 * 264];                 // v transposed [d][s], padded
    __shared__ bf16_t sP[8][512];                   // per-wave P bounce [16][32]

    const int bh = blockIdx.x;
    const int b = bh / Hc, h = bh % Hc;
    const int tb = blockIdx.y;                      // row block 0/1
    const size_t base = ((size_t)b * Tc) * Cc + (size_t)h * HDc;
    const int tid = threadIdx.x, lane = tid & 31, wave = tid >> 5;

    // cooperative load of K (row major) and V (transposed) into LDS
    for (int ch = tid; ch < 2048; ch += 256) {
        int s = ch >> 3, g = (ch & 7) << 3;
        *(uint4*)(&sK[s * 72 + g]) = *(const uint4*)(&kg[base + (size_t)s * Cc + g]);
        uint4 raw = *(const uint4*)(&vg[base + (size_t)s * Cc + g]);
        const bf16_t* e = (const bf16_t*)&raw;
#pragma unroll
        for (int x = 0; x < 8; ++x) sV[(g + x) * 264 + s] = e[x];
    }
    __syncthreads();

    const int t0 = tb * 128 + wave * 16;            // this wave's 16 query rows
    const int kh = (lane >> 4) << 3;
    const int kbo = (lane >> 4) << 3;

    Frag aq[2];                                     // q rows as A-frags, d = 0..63
#pragma unroll
    for (int kb2 = 0; kb2 < 2; ++kb2) {
        const bf16_t* qp = &qg[base + (size_t)(t0 + (lane & 15)) * Cc + kb2 * 32 + kh];
        aq[kb2].q[0] = *(const uint4*)(qp);
        aq[kb2].q[1] = *(const uint4*)(qp + 16);
    }

    const v8f vzero = {0.f, 0.f, 0.f, 0.f, 0.f, 0.f, 0.f, 0.f};
    v8f oacc[4];
    float mrun[8], lrun[8];
#pragma unroll
    for (int j = 0; j < 4; ++j) oacc[j] = vzero;
#pragma unroll
    for (int r = 0; r < 8; ++r) { mrun[r] = -__builtin_inff(); lrun[r] = 0.f; }

    const float scale = 0.044194173824159216f;      // 1/sqrt(C) per reference
    const int nblk = (t0 + 15) / 32 + 1;            // causal: skip fully-masked 32-blocks

    for (int jb = 0; jb < nblk; ++jb) {
        float sv[2][8];
#pragma unroll
        for (int jn = 0; jn < 2; ++jn) {            // two 16-col score tiles
            int s0 = jb * 32 + jn * 16;
            int n = s0 + (lane & 15);
            Frag bk0, bk1;
            bk0.q[0] = *(const uint4*)(&sK[n * 72 + kbo]);
            bk0.q[1] = *(const uint4*)(&sK[n * 72 + kbo + 16]);
            bk1.q[0] = *(const uint4*)(&sK[n * 72 + 32 + kbo]);
            bk1.q[1] = *(const uint4*)(&sK[n * 72 + 32 + kbo + 16]);
            v8f sc = vzero;
            sc = wmma_bf16(aq[0], bk0, sc);
            sc = wmma_bf16(aq[1], bk1, sc);
#pragma unroll
            for (int r = 0; r < 8; ++r) {
                int t = t0 + r + ((lane >> 4) << 3);
                int s = s0 + (lane & 15);
                sv[jn][r] = (s <= t) ? sc[r] * scale : -__builtin_inff();
            }
        }
        // online softmax update (row stats replicated across each 16-lane half)
#pragma unroll
        for (int r = 0; r < 8; ++r) {
            float mx = fmaxf(sv[0][r], sv[1][r]);
#pragma unroll
            for (int d = 1; d < 16; d <<= 1) mx = fmaxf(mx, __shfl_xor(mx, d, 32));
            float mnew = fmaxf(mrun[r], mx);
            float alpha = __expf(mrun[r] - mnew);
            mrun[r] = mnew;
            float p0 = __expf(sv[0][r] - mnew);
            float p1 = __expf(sv[1][r] - mnew);
            sv[0][r] = p0; sv[1][r] = p1;
            float ps = p0 + p1;
#pragma unroll
            for (int d = 1; d < 16; d <<= 1) ps += __shfl_xor(ps, d, 32);
            lrun[r] = lrun[r] * alpha + ps;
#pragma unroll
            for (int j = 0; j < 4; ++j) oacc[j][r] *= alpha;
        }
        // bounce P through per-wave LDS: C-frag layout -> A-frag layout
        bf16_t* pw = &sP[wave][0];
#pragma unroll
        for (int jn = 0; jn < 2; ++jn)
#pragma unroll
            for (int r = 0; r < 8; ++r) {
                int m = r + ((lane >> 4) << 3);
                pw[m * 32 + jn * 16 + (lane & 15)] = (bf16_t)sv[jn][r];
            }
        Frag pa;
        {
            int m = lane & 15;
            pa.q[0] = *(const uint4*)(&pw[m * 32 + kh]);
            pa.q[1] = *(const uint4*)(&pw[m * 32 + kh + 16]);
        }
        // o += P[16x32] * V[32x64]
#pragma unroll
        for (int j = 0; j < 4; ++j) {
            Frag bv;
            int n = j * 16 + (lane & 15);
            int so = jb * 32 + kbo;
            bv.q[0] = *(const uint4*)(&sV[n * 264 + so]);
            bv.q[1] = *(const uint4*)(&sV[n * 264 + so + 16]);
            oacc[j] = wmma_bf16(pa, bv, oacc[j]);
        }
    }

    // normalize and store o (bf16) back at [b, t, h*64 + d]
#pragma unroll
    for (int j = 0; j < 4; ++j)
#pragma unroll
        for (int r = 0; r < 8; ++r) {
            int t = t0 + r + ((lane >> 4) << 3);
            int n = j * 16 + (lane & 15);
            og[base + (size_t)t * Cc + n] = (bf16_t)(oacc[j][r] / lrun[r]);
        }
}

// ---------------- host launcher ----------------
extern "C" void kernel_launch(void* const* d_in, const int* in_sizes, int n_in,
                              void* d_out, int out_size, void* d_ws, size_t ws_size,
                              hipStream_t stream) {
    const int*   idx  = (const int*)  d_in[0];
    const float* tokE = (const float*)d_in[1];
    const float* posE = (const float*)d_in[2];
    const float* Wq   = (const float*)d_in[3];
    const float* Wk   = (const float*)d_in[4];
    const float* Wv   = (const float*)d_in[5];
    const float* Wo   = (const float*)d_in[6];
    const float* bo   = (const float*)d_in[7];
    const float* ln1g = (const float*)d_in[8];
    const float* ln1b = (const float*)d_in[9];
    const float* ln2g = (const float*)d_in[10];
    const float* ln2b = (const float*)d_in[11];
    const float* W1   = (const float*)d_in[12];
    const float* b1   = (const float*)d_in[13];
    const float* W2   = (const float*)d_in[14];
    const float* b2   = (const float*)d_in[15];
    const float* Wlm  = (const float*)d_in[16];
    const float* blm  = (const float*)d_in[17];
    float* out = (float*)d_out;

    char* wsp = (char*)d_ws;
    size_t off = 0;
    auto carve = [&](size_t bytes) -> void* {
        void* p = wsp + off;
        off += (bytes + 255) & ~(size_t)255;
        return p;
    };
    float*  X   = (float*) carve((size_t)BTc * Cc * 4);      // residual stream (f32)
    bf16_t* Hb  = (bf16_t*)carve((size_t)BTc * Cc * 2);      // ln out / attn out (bf16)
    bf16_t* Qb  = (bf16_t*)carve((size_t)BTc * Cc * 2);
    bf16_t* Kb  = (bf16_t*)carve((size_t)BTc * Cc * 2);
    bf16_t* Vb  = (bf16_t*)carve((size_t)BTc * Cc * 2);
    bf16_t* Fb  = (bf16_t*)carve((size_t)BTc * FFc * 2);     // ffn hidden (bf16)
    bf16_t* Wqt = (bf16_t*)carve((size_t)Cc * Cc * 2);       // per-layer bf16 weights
    bf16_t* Wkt = (bf16_t*)carve((size_t)Cc * Cc * 2);
    bf16_t* Wvt = (bf16_t*)carve((size_t)Cc * Cc * 2);
    bf16_t* Wot = (bf16_t*)carve((size_t)Cc * Cc * 2);
    bf16_t* W1t = (bf16_t*)carve((size_t)FFc * Cc * 2);
    bf16_t* W2t = (bf16_t*)carve((size_t)FFc * Cc * 2);
    bf16_t* Wlt = (bf16_t*)carve((size_t)VPADc * Cc * 2);
    (void)in_sizes; (void)n_in; (void)out_size; (void)ws_size;

    embed_kernel<<<BTc, 256, 0, stream>>>(idx, tokE, posE, X);
    conv_wt<<<(VPADc * Cc) / 256, 256, 0, stream>>>(Wlm, Wlt, Cc, Vc, VPADc);

    const dim3 gC(Cc / 128, BTc / 128);             // N=512 GEMMs
    for (int l = 0; l < Lc; ++l) {
        size_t wOff = (size_t)l * Cc * Cc;
        conv_qkv<<<(Cc * Cc) / 256, 256, 0, stream>>>(Wq + wOff, Wqt);
        conv_qkv<<<(Cc * Cc) / 256, 256, 0, stream>>>(Wk + wOff, Wkt);
        conv_qkv<<<(Cc * Cc) / 256, 256, 0, stream>>>(Wv + wOff, Wvt);
        conv_wt<<<(Cc * Cc) / 256, 256, 0, stream>>>(Wo + wOff, Wot, Cc, Cc, Cc);
        conv_wt<<<(FFc * Cc) / 256, 256, 0, stream>>>(W1 + (size_t)l * Cc * FFc, W1t, Cc, FFc, FFc);
        conv_wt<<<(Cc * FFc) / 256, 256, 0, stream>>>(W2 + (size_t)l * FFc * Cc, W2t, FFc, Cc, Cc);

        ln_kernel<<<BTc, 256, 0, stream>>>(X, ln1g + l * Cc, ln1b + l * Cc, Hb);
        gemm_bf16<EPI_BF16><<<gC, 256, 0, stream>>>(Hb, Wqt, nullptr, nullptr, nullptr, Qb, BTc, Cc, Cc, 0, 0);
        gemm_bf16<EPI_BF16><<<gC, 256, 0, stream>>>(Hb, Wkt, nullptr, nullptr, nullptr, Kb, BTc, Cc, Cc, 0, 0);
        gemm_bf16<EPI_BF16><<<gC, 256, 0, stream>>>(Hb, Wvt, nullptr, nullptr, nullptr, Vb, BTc, Cc, Cc, 0, 0);

        attn_kernel<<<dim3(Bc * Hc, 2), 256, 0, stream>>>(Qb, Kb, Vb, Hb);

        gemm_bf16<EPI_RES_F32><<<gC, 256, 0, stream>>>(Hb, Wot, bo + l * Cc, X, X, nullptr, BTc, Cc, Cc, 0, 0);

        ln_kernel<<<BTc, 256, 0, stream>>>(X, ln2g + l * Cc, ln2b + l * Cc, Hb);
        gemm_bf16<EPI_BF16_RELU><<<dim3(FFc / 128, BTc / 128), 256, 0, stream>>>(
            Hb, W1t, b1 + (size_t)l * FFc, nullptr, nullptr, Fb, BTc, FFc, Cc, 0, 0);
        gemm_bf16<EPI_RES_F32><<<gC, 256, 0, stream>>>(Fb, W2t, b2 + l * Cc, X, X, nullptr, BTc, Cc, FFc, 0, 0);
    }

    cast_kernel<<<(BTc * Cc) / 256, 256, 0, stream>>>(X, Hb);
    gemm_bf16<EPI_F32_MASK><<<dim3(1, BTc / 128), 256, 0, stream>>>(
        Hb, Wlt, blm, nullptr, out, nullptr, BTc, VPADc, Cc, Vc, Vc);
}
